// DeepHeuristicHypergraphAttention_29815662969093
// MI455X (gfx1250) — matrix operand order repaired and checked by reading, not compiled, not run on hardware
//
#include <hip/hip_runtime.h>
#include <hip/hip_bf16.h>
#include <math.h>

// ---------------------------------------------------------------------------
// CDNA5 (gfx1250) WMMA types
// ---------------------------------------------------------------------------
typedef __attribute__((ext_vector_type(16))) __bf16 v16bf;
typedef __attribute__((ext_vector_type(8)))  float  v8f;

#define LN_EPS 1e-5f

struct alignas(16) U4 { unsigned int x, y, z, w; };
union FragB16 { v16bf v; U4 q[2]; };

static __device__ __forceinline__ unsigned short f2bf(float f) {
  unsigned int u = __float_as_uint(f);
  u += 0x7FFFu + ((u >> 16) & 1u);   // round-to-nearest-even
  return (unsigned short)(u >> 16);
}
static __device__ __forceinline__ float bf2f(unsigned short s) {
  return __uint_as_float(((unsigned int)s) << 16);
}
static __device__ __forceinline__ float gelu_exact(float x) {
  return 0.5f * x * (1.0f + erff(x * 0.70710678118654752f));
}
static __device__ __forceinline__ float sigmoid_(float x) {
  return 1.0f / (1.0f + expf(-x));
}
static __device__ __forceinline__ float wave_red_add(float v) {
  #pragma unroll
  for (int m = 16; m >= 1; m >>= 1) v += __shfl_xor(v, m, 32);
  return v;
}

// gfx1250 async global->LDS copy (ASYNCcnt-tracked), per CDNA5 ISA.
// LDS byte address = low 32 bits of the generic shared pointer (ISA 10.2 aperture rule).
static __device__ __forceinline__ void async_copy_b128(unsigned lds_off, const void* gptr) {
  asm volatile("global_load_async_to_lds_b128 %0, %1, off"
               :: "v"(lds_off), "v"(gptr) : "memory");
}
static __device__ __forceinline__ void wait_asynccnt0() {
  asm volatile("s_wait_asynccnt 0" ::: "memory");
}

// ---------------------------------------------------------------------------
// Pack a row-major fp32 weight [K,N] into bf16 WMMA B-fragment order:
// packed[((kt*ntiles+nt)*32 + lane)*16 + t] = W[kt*32 + (t>>3)*16 + (lane>>4)*8 + (t&7)][nt*16 + (lane&15)]
// so each lane's 16 bf16 B-fragment values are one contiguous 32-byte chunk.
// ---------------------------------------------------------------------------
__global__ void pack_b_kernel(const float* __restrict__ W, unsigned short* __restrict__ out,
                              int K, int N) {
  int p = blockIdx.x * blockDim.x + threadIdx.x;
  if (p >= K * N) return;
  int ntiles = N >> 4;
  int t    = p & 15;
  int lane = (p >> 4) & 31;
  int tile = p >> 9;
  int nt   = tile % ntiles;
  int kt   = tile / ntiles;
  int k = kt * 32 + ((t >> 3) << 4) + ((lane >> 4) << 3) + (t & 7);
  int n = nt * 16 + (lane & 15);
  out[p] = f2bf(W[(size_t)k * N + n]);
}

__global__ void cvt_bf16_kernel(const float* __restrict__ in, unsigned short* __restrict__ out, int n) {
  int i = blockIdx.x * blockDim.x + threadIdx.x;
  if (i < n) out[i] = f2bf(in[i]);
}

// ---------------------------------------------------------------------------
// Fused: node_features = LN(ent_embs + role_emb[roles]) -> bf16 row-major.
// Also precomputes gate(b,n) = sigmoid(log1p(idf)*gw + gb) * mask.
// One block of 256 per row of 768.
// ---------------------------------------------------------------------------
__global__ __launch_bounds__(256)
void node_ln_kernel(const float* __restrict__ ent, const int* __restrict__ roles,
                    const float* __restrict__ role_emb,
                    const float* __restrict__ lng, const float* __restrict__ lnb,
                    const float* __restrict__ idfs, const float* __restrict__ mask,
                    const float* __restrict__ gate_w, const float* __restrict__ gate_b,
                    unsigned short* __restrict__ nf, float* __restrict__ gate) {
  __shared__ float red[16];
  __shared__ float stats[2];
  const int row = blockIdx.x;
  const int tid = threadIdx.x;
  const int role = roles[row];
  const float* e  = ent + (size_t)row * 768;
  const float* re = role_emb + (size_t)role * 768;
  float v[3]; float s = 0.f, s2 = 0.f;
  #pragma unroll
  for (int j = 0; j < 3; ++j) {
    int c = tid + j * 256;
    float x = e[c] + re[c];
    v[j] = x; s += x; s2 += x * x;
  }
  s = wave_red_add(s); s2 = wave_red_add(s2);
  int wave = tid >> 5, lane = tid & 31;
  if (lane == 0) { red[wave] = s; red[8 + wave] = s2; }
  __syncthreads();
  if (tid == 0) {
    float ts = 0.f, ts2 = 0.f;
    #pragma unroll
    for (int w = 0; w < 8; ++w) { ts += red[w]; ts2 += red[8 + w]; }
    float mu  = ts * (1.0f / 768.0f);
    float var = ts2 * (1.0f / 768.0f) - mu * mu;
    stats[0] = mu; stats[1] = rsqrtf(var + LN_EPS);
    gate[row] = sigmoid_(log1pf(idfs[row]) * gate_w[0] + gate_b[0]) * mask[row];
  }
  __syncthreads();
  float mu = stats[0], rstd = stats[1];
  #pragma unroll
  for (int j = 0; j < 3; ++j) {
    int c = tid + j * 256;
    nf[(size_t)row * 768 + c] = f2bf((v[j] - mu) * rstd * lng[c] + lnb[c]);
  }
}

// ---------------------------------------------------------------------------
// bf16 WMMA GEMM + bias + exact-erf gelu -> bf16:
//   C[M,N] = gelu(A[M,K] @ B[K,N] + bias)
// Block = 256 threads (8 waves), covers 32 rows x 512 cols of C.
// The 32xK A slab is staged once into LDS via gfx1250 async global->LDS DMA
// (shared by all 8 waves); each wave register-blocks 2 M-tiles x 4 N-tiles
// (8 WMMAs per K-step vs 4 LDS + 8 global b128 loads).
// Grid: (M/32, N/512).
// ---------------------------------------------------------------------------
#define LDS_AP 776   // padded A-slab row stride in bf16 elems (1552 B: 16B-aligned, bank-skewed)

__global__ __launch_bounds__(256)
void gemm_gelu_kernel(const unsigned short* __restrict__ A,
                      const unsigned short* __restrict__ pB,
                      const float* __restrict__ bias,
                      unsigned short* __restrict__ C,
                      int M, int K, int N) {
  __shared__ unsigned short ash[32 * LDS_AP];

  const int tid   = threadIdx.x;
  const int wave  = tid >> 5;
  const int lane  = tid & 31;
  const int mrow0 = blockIdx.x * 32;
  const int ntiles = N >> 4;
  const int ktiles = K >> 5;

  // ---- async-stage the 32xK A slab into LDS ----
  const int cpr = K >> 3;                      // 16-byte chunks per row
  for (int idx = tid; idx < 32 * cpr; idx += 256) {
    int row = idx / cpr, cc = idx - row * cpr;
    unsigned loff = (unsigned)(uintptr_t)&ash[row * LDS_AP + cc * 8];
    async_copy_b128(loff, (const void*)(A + (size_t)(mrow0 + row) * K + cc * 8));
  }
  wait_asynccnt0();
  __syncthreads();

  const int koff = (lane >> 4) << 3;           // A-fragment half select
  const int arow = lane & 15;
  const int ntb  = blockIdx.y * 32 + wave * 4; // first of this wave's 4 N-tiles

  v8f zero8 = {0.f, 0.f, 0.f, 0.f, 0.f, 0.f, 0.f, 0.f};
  v8f acc[2][4];
  #pragma unroll
  for (int m = 0; m < 2; ++m)
    #pragma unroll
    for (int j = 0; j < 4; ++j) acc[m][j] = zero8;

  for (int kt = 0; kt < ktiles; ++kt) {
    const int kb = kt * 32 + koff;
    FragB16 a0, a1;
    a0.q[0] = *(const U4*)&ash[arow * LDS_AP + kb];
    a0.q[1] = *(const U4*)&ash[arow * LDS_AP + kb + 16];
    a1.q[0] = *(const U4*)&ash[(16 + arow) * LDS_AP + kb];
    a1.q[1] = *(const U4*)&ash[(16 + arow) * LDS_AP + kb + 16];
    const unsigned short* bbase = pB + ((size_t)kt * ntiles + ntb) * 512 + (size_t)lane * 16;
    // WGP-scope prefetch of next K-step's B tiles (locality=3 -> near scope,
    // pulls into all cache levels per the gfx1250 prefetch-TH table)
    __builtin_prefetch(bbase + (size_t)ntiles * 512, 0, 3);
    #pragma unroll
    for (int j = 0; j < 4; ++j) {
      FragB16 bf;
      const unsigned short* bp = bbase + (size_t)j * 512;
      bf.q[0] = *(const U4*)(bp);
      bf.q[1] = *(const U4*)(bp + 8);
      acc[0][j] = __builtin_amdgcn_wmma_f32_16x16x32_bf16(false, a0.v, false, bf.v,
                                                          (short)0, acc[0][j], false, false);
      acc[1][j] = __builtin_amdgcn_wmma_f32_16x16x32_bf16(false, a1.v, false, bf.v,
                                                          (short)0, acc[1][j], false, false);
    }
  }

  #pragma unroll
  for (int m = 0; m < 2; ++m) {
    const int mb = mrow0 + m * 16 + ((lane >> 4) << 3);
    #pragma unroll
    for (int j = 0; j < 4; ++j) {
      const int ncol = (ntb + j) * 16 + (lane & 15);
      const float bv = bias[ncol];
      unsigned short* cp = C + (size_t)mb * N + ncol;
      #pragma unroll
      for (int i = 0; i < 8; ++i)
        cp[(size_t)i * N] = f2bf(gelu_exact(acc[m][j][i] + bv));
    }
  }
}

// ---------------------------------------------------------------------------
// Fused per-row MLP: builds interaction rows [q,k,|q-k|,q*k] (K=512) for 16
// rows in LDS, then W1-WMMA (N=256) + gelu + LN -> W2-WMMA (N=128) + gelu + LN
// -> W3 dot -> gate -> atomic accumulate into per-(b,h) sums.
// Block = 512 threads = 16 waves; grid = 65536/16 = 4096.
// Row index r = (b*128 + n)*8 + h.
// ---------------------------------------------------------------------------
__global__ __launch_bounds__(512)
void mlp_score_kernel(const unsigned short* __restrict__ qbuf,   // [64,1024] bf16
                      const unsigned short* __restrict__ kbuf,   // [8192,1024] bf16
                      const unsigned short* __restrict__ pW1,    // packed [512,256]
                      const float* __restrict__ b1,
                      const float* __restrict__ ln1g, const float* __restrict__ ln1b,
                      const unsigned short* __restrict__ pW2,    // packed [256,128]
                      const float* __restrict__ b2,
                      const float* __restrict__ ln2g, const float* __restrict__ ln2b,
                      const float* __restrict__ W3, const float* __restrict__ b3,
                      const float* __restrict__ gate,            // [8192]
                      float* __restrict__ acc_out) {             // [64*8]
  __shared__ unsigned short xsh[16][520];   // interaction rows, bf16
  __shared__ float          h1f[16][260];   // layer-1 pre-LN fp32
  __shared__ unsigned short h1bf[16][264];  // layer-1 post-LN bf16 (A for layer 2)
  __shared__ float          h2f[16][132];   // layer-2 pre-LN fp32

  const int tid  = threadIdx.x;
  const int wave = tid >> 5;
  const int lane = tid & 31;
  const int r0   = blockIdx.x * 16;

  // ---- build interaction rows into LDS ----
  for (int idx = tid; idx < 16 * 512; idx += 512) {
    int row = idx >> 9, col = idx & 511;
    int r = r0 + row;
    int b = r >> 10, rem = r & 1023;
    int n = rem >> 3, h = rem & 7;
    int c = col & 127;
    float qv = bf2f(qbuf[((size_t)b << 10) + (h << 7) + c]);
    float kv = bf2f(kbuf[(size_t)((b << 7) + n) * 1024 + (h << 7) + c]);
    int seg = col >> 7;
    float v = (seg == 0) ? qv : (seg == 1) ? kv : (seg == 2) ? fabsf(qv - kv) : qv * kv;
    xsh[row][col] = f2bf(v);
  }
  __syncthreads();

  const int koff = (lane >> 4) << 3;
  const int arow = lane & 15;

  // ---- layer 1: 16 waves, wave w -> N-tile w (256/16 tiles), K = 512 ----
  {
    const int nt = wave;
    v8f acc = {0.f, 0.f, 0.f, 0.f, 0.f, 0.f, 0.f, 0.f};
    for (int kt = 0; kt < 16; ++kt) {
      FragB16 af, bf;
      int kb = kt * 32 + koff;
      af.q[0] = *(const U4*)&xsh[arow][kb];
      af.q[1] = *(const U4*)&xsh[arow][kb + 16];
      const unsigned short* bp = pW1 + ((size_t)(kt * 16 + nt) * 32 + lane) * 16;
      bf.q[0] = *(const U4*)(bp);
      bf.q[1] = *(const U4*)(bp + 8);
      acc = __builtin_amdgcn_wmma_f32_16x16x32_bf16(false, af.v, false, bf.v,
                                                    (short)0, acc, false, false);
    }
    const int ncol = nt * 16 + (lane & 15);
    const float bv = b1[ncol];
    const int mb = (lane >> 4) << 3;
    #pragma unroll
    for (int i = 0; i < 8; ++i)
      h1f[mb + i][ncol] = gelu_exact(acc[i] + bv);
  }
  __syncthreads();

  // ---- LN over 256 per row; wave w handles row w (8 cols/lane) ----
  {
    const int row = wave;
    float vals[8]; float s = 0.f, s2 = 0.f;
    #pragma unroll
    for (int j = 0; j < 8; ++j) {
      float v = h1f[row][lane * 8 + j];
      vals[j] = v; s += v; s2 += v * v;
    }
    s = wave_red_add(s); s2 = wave_red_add(s2);
    float mu = s * (1.0f / 256.0f);
    float rstd = rsqrtf(s2 * (1.0f / 256.0f) - mu * mu + LN_EPS);
    #pragma unroll
    for (int j = 0; j < 8; ++j) {
      int c = lane * 8 + j;
      h1bf[row][c] = f2bf((vals[j] - mu) * rstd * ln1g[c] + ln1b[c]);
    }
  }
  __syncthreads();

  // ---- layer 2: waves 0..7, wave w -> N-tile w (128/16 tiles), K = 256 ----
  if (wave < 8) {
    const int nt = wave;
    v8f acc = {0.f, 0.f, 0.f, 0.f, 0.f, 0.f, 0.f, 0.f};
    for (int kt = 0; kt < 8; ++kt) {
      FragB16 af, bf;
      int kb = kt * 32 + koff;
      af.q[0] = *(const U4*)&h1bf[arow][kb];
      af.q[1] = *(const U4*)&h1bf[arow][kb + 16];
      const unsigned short* bp = pW2 + ((size_t)(kt * 8 + nt) * 32 + lane) * 16;
      bf.q[0] = *(const U4*)(bp);
      bf.q[1] = *(const U4*)(bp + 8);
      acc = __builtin_amdgcn_wmma_f32_16x16x32_bf16(false, af.v, false, bf.v,
                                                    (short)0, acc, false, false);
    }
    const int ncol = nt * 16 + (lane & 15);
    const float bv = b2[ncol];
    const int mb = (lane >> 4) << 3;
    #pragma unroll
    for (int i = 0; i < 8; ++i)
      h2f[mb + i][ncol] = gelu_exact(acc[i] + bv);
  }
  __syncthreads();

  // ---- LN over 128 + W3 dot + gate + accumulate; wave w handles row w ----
  {
    const int row = wave;
    float vals[4]; float s = 0.f, s2 = 0.f;
    #pragma unroll
    for (int j = 0; j < 4; ++j) {
      float v = h2f[row][lane * 4 + j];
      vals[j] = v; s += v; s2 += v * v;
    }
    s = wave_red_add(s); s2 = wave_red_add(s2);
    float mu = s * (1.0f / 128.0f);
    float rstd = rsqrtf(s2 * (1.0f / 128.0f) - mu * mu + LN_EPS);
    float part = 0.f;
    #pragma unroll
    for (int j = 0; j < 4; ++j) {
      int c = lane * 4 + j;
      float nv = (vals[j] - mu) * rstd * ln2g[c] + ln2b[c];
      part += nv * W3[c];
    }
    part = wave_red_add(part);
    if (lane == 0) {
      int r = r0 + row;
      int b = r >> 10, rem = r & 1023;
      int n = rem >> 3, h = rem & 7;
      float sc = (part + b3[0]) * gate[(b << 7) + n];
      atomicAdd(&acc_out[(b << 3) + h], sc);
    }
  }
}

__global__ void zero_acc_kernel(float* __restrict__ acc) {
  acc[threadIdx.x] = 0.f;
}

__global__ void finalize_kernel(const float* __restrict__ acc, float* __restrict__ out) {
  int b = threadIdx.x;   // 64 threads
  float s = 0.f;
  #pragma unroll
  for (int h = 0; h < 8; ++h) s += acc[b * 8 + h];
  out[b] = sigmoid_(s * 0.125f);
}

// ---------------------------------------------------------------------------
// Launch
// ---------------------------------------------------------------------------
extern "C" void kernel_launch(void* const* d_in, const int* in_sizes, int n_in,
                              void* d_out, int out_size, void* d_ws, size_t ws_size,
                              hipStream_t stream) {
  const float* q_emb    = (const float*)d_in[0];
  const float* ent      = (const float*)d_in[1];
  const int*   roles    = (const int*)  d_in[2];
  const float* idfs     = (const float*)d_in[3];
  const float* mask     = (const float*)d_in[4];
  const float* role_emb = (const float*)d_in[5];
  const float* ln_f_g   = (const float*)d_in[6];
  const float* ln_f_b   = (const float*)d_in[7];
  const float* Wq       = (const float*)d_in[8];
  const float* bq       = (const float*)d_in[9];
  const float* Wk       = (const float*)d_in[10];
  const float* bk       = (const float*)d_in[11];
  const float* W1       = (const float*)d_in[12];
  const float* b1       = (const float*)d_in[13];
  const float* ln1g     = (const float*)d_in[14];
  const float* ln1b     = (const float*)d_in[15];
  const float* W2       = (const float*)d_in[16];
  const float* b2       = (const float*)d_in[17];
  const float* ln2g     = (const float*)d_in[18];
  const float* ln2b     = (const float*)d_in[19];
  const float* W3       = (const float*)d_in[20];
  const float* b3       = (const float*)d_in[21];
  const float* gate_w   = (const float*)d_in[22];
  const float* gate_b   = (const float*)d_in[23];

  char* ws = (char*)d_ws;
  size_t off = 0;
  auto wsalloc = [&](size_t bytes) -> void* {
    void* p = ws + off;
    off = (off + bytes + 255) & ~(size_t)255;
    return p;
  };
  unsigned short* qe_bf = (unsigned short*)wsalloc((size_t)64 * 768 * 2);
  unsigned short* nf    = (unsigned short*)wsalloc((size_t)8192 * 768 * 2);
  unsigned short* pWq   = (unsigned short*)wsalloc((size_t)768 * 1024 * 2);
  unsigned short* pWk   = (unsigned short*)wsalloc((size_t)768 * 1024 * 2);
  unsigned short* pW1   = (unsigned short*)wsalloc((size_t)512 * 256 * 2);
  unsigned short* pW2   = (unsigned short*)wsalloc((size_t)256 * 128 * 2);
  unsigned short* qbuf  = (unsigned short*)wsalloc((size_t)64 * 1024 * 2);
  unsigned short* kbuf  = (unsigned short*)wsalloc((size_t)8192 * 1024 * 2);
  float*          gate  = (float*)wsalloc((size_t)8192 * 4);
  float*          accb  = (float*)wsalloc((size_t)512 * 4);

  // 1) one-time conversions / weight packing
  cvt_bf16_kernel<<<(64 * 768 + 255) / 256, 256, 0, stream>>>(q_emb, qe_bf, 64 * 768);
  pack_b_kernel<<<(768 * 1024 + 255) / 256, 256, 0, stream>>>(Wq, pWq, 768, 1024);
  pack_b_kernel<<<(768 * 1024 + 255) / 256, 256, 0, stream>>>(Wk, pWk, 768, 1024);
  pack_b_kernel<<<(512 * 256 + 255) / 256, 256, 0, stream>>>(W1, pW1, 512, 256);
  pack_b_kernel<<<(256 * 128 + 255) / 256, 256, 0, stream>>>(W2, pW2, 256, 128);

  // 2) fused role-emb + LN node features (bf16) + gate precompute
  node_ln_kernel<<<8192, 256, 0, stream>>>(ent, roles, role_emb, ln_f_g, ln_f_b,
                                           idfs, mask, gate_w, gate_b, nf, gate);

  // 3) WMMA projections with gelu epilogue (async-LDS A staging, 2x4 reg blocking)
  gemm_gelu_kernel<<<dim3(2, 2),   256, 0, stream>>>(qe_bf, pWq, bq, qbuf, 64,   768, 1024);
  gemm_gelu_kernel<<<dim3(256, 2), 256, 0, stream>>>(nf,    pWk, bk, kbuf, 8192, 768, 1024);

  // 4) fused interaction MLP + scoring
  zero_acc_kernel<<<1, 512, 0, stream>>>(accb);
  mlp_score_kernel<<<4096, 512, 0, stream>>>(qbuf, kbuf, pW1, b1, ln1g, ln1b,
                                             pW2, b2, ln2g, ln2b, W3, b3, gate, accb);

  // 5) mean over heads + sigmoid
  finalize_kernel<<<1, 64, 0, stream>>>(accb, (float*)d_out);
}
